// _GQAttention_6682969112858
// MI455X (gfx1250) — matrix operand order repaired and checked by reading, not compiled, not run on hardware
//
#include <hip/hip_runtime.h>
#include <hip/hip_bf16.h>

// ---------------- CDNA5 WMMA types/helpers ----------------
typedef __attribute__((ext_vector_type(16))) _Float16 v16h;
typedef __attribute__((ext_vector_type(8)))  float    v8f;

#define WMMA16(a, b, c) \
  __builtin_amdgcn_wmma_f32_16x16x32_f16(false, (a), false, (b), (short)0, (c), false, false)

// Problem constants (match reference)
constexpr int Bv   = 4;
constexpr int Tv   = 2048;
constexpr int Dm   = 1024;
constexpr int HQ   = 16;
constexpr int HKV  = 4;
constexpr int HD   = 64;
constexpr int KVD  = 2 * HKV * HD;   // 512
constexpr int Mrows = Bv * Tv;       // 8192

// Fragment loads from per-lane pre-offset pointers (ISA 7.12.2 layouts).
// A (16x32, MxK): lane base = A + (lane&15)*lda + (lane<16?0:8); halves at +0..7 and +16..23.
__device__ __forceinline__ v16h frag_a_at(const _Float16* p) {
  v16h a;
#pragma unroll
  for (int i = 0; i < 8; ++i) a[i] = p[i];
#pragma unroll
  for (int i = 0; i < 8; ++i) a[8 + i] = p[16 + i];
  return a;
}
// B (32x16, KxN) from column-major BT[N][ldb]: lane base = BT + (lane&15)*ldb + (lane<16?0:8);
// halves at +0..7 and +16..23.
__device__ __forceinline__ v16h frag_b_at(const _Float16* p) {
  v16h b;
#pragma unroll
  for (int i = 0; i < 8; ++i) b[i] = p[i];
#pragma unroll
  for (int i = 0; i < 8; ++i) b[8 + i] = p[16 + i];
  return b;
}

__device__ __forceinline__ const _Float16* lane_base_a(const _Float16* base, int lda, int lane) {
  return base + (lane & 15) * lda + ((lane < 16) ? 0 : 8);
}
__device__ __forceinline__ const _Float16* lane_base_b(const _Float16* baseT, int ldb, int lane) {
  return baseT + (long)(lane & 15) * ldb + ((lane < 16) ? 0 : 8);
}

// ---------------- Conversion / transpose kernels ----------------
__global__ void k_f32_to_f16(const float* __restrict__ src, _Float16* __restrict__ dst, int n) {
  int i = blockIdx.x * blockDim.x + threadIdx.x;
  if (i < n) dst[i] = (_Float16)src[i];
}

// W[K][N] f32 -> WT[N][K] f16
__global__ void k_transpose_w(const float* __restrict__ W, _Float16* __restrict__ WT, int K, int N) {
  int i = blockIdx.x * blockDim.x + threadIdx.x;
  if (i >= K * N) return;
  int k = i / N, n = i % N;
  WT[(long)n * K + k] = (_Float16)W[i];
}

// ---------------- WMMA GEMM: C[M][N] = A[M][K] x BT[N][K]^T ----------------
// One wave computes a 32(M) x 64(N) strip: 8 C tiles.
// k-loop unrolled by 64 with ping-pong fragment sets (no rotation copies).
// Requires K % 64 == 0.
__global__ void __launch_bounds__(256)
k_gemm_wmma(const _Float16* __restrict__ A, const _Float16* __restrict__ BT,
            _Float16* __restrict__ C16, float* __restrict__ C32,
            int M, int N, int K, int out16) {
  int lane = threadIdx.x & 31;
  int wave = threadIdx.x >> 5;
  long strip = (long)blockIdx.x * (blockDim.x >> 5) + wave;
  int stripsN = N >> 6;                // N/64
  int m0 = (int)(strip / stripsN) * 32;
  int n0 = (int)(strip % stripsN) * 64;
  if (m0 >= M) return;

  // hoisted per-lane base pointers
  const _Float16* pa0 = lane_base_a(A + (long)m0 * K, K, lane);
  const _Float16* pa1 = lane_base_a(A + (long)(m0 + 16) * K, K, lane);
  const _Float16* pb0 = lane_base_b(BT + (long)(n0 + 0)  * K, K, lane);
  const _Float16* pb1 = lane_base_b(BT + (long)(n0 + 16) * K, K, lane);
  const _Float16* pb2 = lane_base_b(BT + (long)(n0 + 32) * K, K, lane);
  const _Float16* pb3 = lane_base_b(BT + (long)(n0 + 48) * K, K, lane);

  v8f c[8] = {v8f{}, v8f{}, v8f{}, v8f{}, v8f{}, v8f{}, v8f{}, v8f{}};

  // prologue: set0 @ k=0
  v16h xa0 = frag_a_at(pa0), xa1 = frag_a_at(pa1);
  v16h xb0 = frag_b_at(pb0), xb1 = frag_b_at(pb1), xb2 = frag_b_at(pb2), xb3 = frag_b_at(pb3);

  for (int kk = 0; kk < K; kk += 64) {
    int k1 = kk + 32;
    int k2 = (kk + 64 < K) ? (kk + 64) : 0;   // dummy wrap reload on final iter

    // L2 prefetch of the A stream ahead (lowers to global_prefetch_b8)
    __builtin_prefetch(pa0 + kk + 512, 0, 1);
    __builtin_prefetch(pa1 + kk + 512, 0, 1);

    // load set1 @ k1 while computing set0
    v16h ya0 = frag_a_at(pa0 + k1), ya1 = frag_a_at(pa1 + k1);
    v16h yb0 = frag_b_at(pb0 + k1), yb1 = frag_b_at(pb1 + k1);
    v16h yb2 = frag_b_at(pb2 + k1), yb3 = frag_b_at(pb3 + k1);

    c[0] = WMMA16(xa0, xb0, c[0]);  c[4] = WMMA16(xa1, xb0, c[4]);
    c[1] = WMMA16(xa0, xb1, c[1]);  c[5] = WMMA16(xa1, xb1, c[5]);
    c[2] = WMMA16(xa0, xb2, c[2]);  c[6] = WMMA16(xa1, xb2, c[6]);
    c[3] = WMMA16(xa0, xb3, c[3]);  c[7] = WMMA16(xa1, xb3, c[7]);

    // load set0 @ k2 while computing set1
    xa0 = frag_a_at(pa0 + k2);  xa1 = frag_a_at(pa1 + k2);
    xb0 = frag_b_at(pb0 + k2);  xb1 = frag_b_at(pb1 + k2);
    xb2 = frag_b_at(pb2 + k2);  xb3 = frag_b_at(pb3 + k2);

    c[0] = WMMA16(ya0, yb0, c[0]);  c[4] = WMMA16(ya1, yb0, c[4]);
    c[1] = WMMA16(ya0, yb1, c[1]);  c[5] = WMMA16(ya1, yb1, c[5]);
    c[2] = WMMA16(ya0, yb2, c[2]);  c[6] = WMMA16(ya1, yb2, c[6]);
    c[3] = WMMA16(ya0, yb3, c[3]);  c[7] = WMMA16(ya1, yb3, c[7]);
  }

  int col = lane & 15, rb = (lane < 16) ? 0 : 8;
#pragma unroll
  for (int half = 0; half < 2; ++half)
#pragma unroll
    for (int j = 0; j < 4; ++j)
#pragma unroll
      for (int i = 0; i < 8; ++i) {
        long m = m0 + half * 16 + rb + i, n = n0 + j * 16 + col;
        float v = c[half * 4 + j][i];
        if (out16) C16[m * N + n] = (_Float16)v;
        else       C32[m * N + n] = v;
      }
}

// ---------------- RoPE + layout kernels ----------------
// q16: [b][t][HQ][64] -> qr: [b][h][t][64] with RoPE
__global__ void k_rope_q(const _Float16* __restrict__ q16, const float* __restrict__ cosv,
                         const float* __restrict__ sinv, _Float16* __restrict__ qr) {
  int i = blockIdx.x * blockDim.x + threadIdx.x;
  if (i >= Bv * Tv * HQ * (HD / 2)) return;
  int p = i & 31;            int r = i >> 5;
  int h = r % HQ;            r /= HQ;
  int t = r % Tv;            int b = r / Tv;
  int d0 = 2 * p;
  float c = cosv[t * HD + d0], s = sinv[t * HD + d0];
  long src = (((long)b * Tv + t) * HQ + h) * HD + d0;
  float x0 = (float)q16[src], x1 = (float)q16[src + 1];
  long dst = (((long)b * HQ + h) * Tv + t) * HD + d0;
  qr[dst]     = (_Float16)(x0 * c - x1 * s);
  qr[dst + 1] = (_Float16)(x1 * c + x0 * s);
}

// kv16: [b][t][HKV][2][64] -> kr: [b][kh][t][64] (RoPE); vt: [b][kh][64][t] (transposed)
__global__ void k_rope_kv(const _Float16* __restrict__ kv16, const float* __restrict__ cosv,
                          const float* __restrict__ sinv, _Float16* __restrict__ kr,
                          _Float16* __restrict__ vt) {
  int i = blockIdx.x * blockDim.x + threadIdx.x;
  if (i >= Bv * Tv * HKV * (HD / 2)) return;
  int p = i & 31;            int r = i >> 5;
  int kh = r % HKV;          r /= HKV;
  int t = r % Tv;            int b = r / Tv;
  int d0 = 2 * p;
  float c = cosv[t * HD + d0], s = sinv[t * HD + d0];
  long src = ((((long)b * Tv + t) * HKV + kh) * 2) * HD;
  float k0 = (float)kv16[src + d0], k1 = (float)kv16[src + d0 + 1];
  float v0 = (float)kv16[src + HD + d0], v1 = (float)kv16[src + HD + d0 + 1];
  long kd = (((long)b * HKV + kh) * Tv + t) * HD + d0;
  kr[kd]     = (_Float16)(k0 * c - k1 * s);
  kr[kd + 1] = (_Float16)(k1 * c + k0 * s);
  long vd = (((long)b * HKV + kh) * HD + d0) * Tv + t;
  vt[vd]       = (_Float16)v0;
  vt[vd + Tv]  = (_Float16)v1;
}

// ---------------- Flash attention (one wave per (b,h,16-query tile)) ----------------
__global__ void __launch_bounds__(32)
k_flash_attn(const _Float16* __restrict__ qr, const _Float16* __restrict__ kr,
             const _Float16* __restrict__ vt, _Float16* __restrict__ att) {
  int lane = threadIdx.x;
  int idx = blockIdx.x;
  int q0 = (idx % (Tv / 16)) * 16;  idx /= (Tv / 16);
  int h  = idx % HQ;                int b = idx / HQ;
  int kh = h / (HQ / HKV);

  const _Float16* Qb = qr + (((long)b * HQ + h) * Tv + q0) * HD;
  const _Float16* Kb = kr + (((long)b * HKV + kh) * Tv) * HD;
  const _Float16* Vb = vt + (((long)b * HKV + kh) * HD) * (long)Tv;

  __shared__ float    Sb[16][32];
  __shared__ _Float16 Pb[16][32];
  __shared__ float    mrow[16], lrow[16], arow[16];

  // hoisted per-lane bases
  const _Float16* pq  = lane_base_a(Qb, HD, lane);
  const _Float16* pkb = lane_base_b(Kb, HD, lane);   // + (key)*HD + dchunk
  const _Float16* pvb = lane_base_b(Vb, Tv, lane);   // + (dchunk rows)*Tv + j
  const _Float16* ppb = lane_base_a(&Pb[0][0], 32, lane);

  // Q fragments are loop-invariant (d-chunks 0..31, 32..63)
  v16h aq0 = frag_a_at(pq);
  v16h aq1 = frag_a_at(pq + 32);

  if (lane < 16) { mrow[lane] = -1e30f; lrow[lane] = 0.0f; }
  __syncthreads();

  v8f o[4] = {v8f{}, v8f{}, v8f{}, v8f{}};
  int col = lane & 15, rb = (lane < 16) ? 0 : 8;

  for (int j = 0; j < Tv; j += 32) {
    // S = (Q K^T) * scale : two 16x16 tiles
#pragma unroll
    for (int sub = 0; sub < 2; ++sub) {
      v8f s = v8f{};
      v16h bk0 = frag_b_at(pkb + (long)(j + sub * 16) * HD + 0);
      s = WMMA16(aq0, bk0, s);
      v16h bk1 = frag_b_at(pkb + (long)(j + sub * 16) * HD + 32);
      s = WMMA16(aq1, bk1, s);
#pragma unroll
      for (int i = 0; i < 8; ++i) Sb[rb + i][sub * 16 + col] = s[i] * 0.125f;
    }
    __syncthreads();

    // online softmax: one lane per query row
    if (lane < 16) {
      float mo = mrow[lane], mn = mo;
#pragma unroll
      for (int k = 0; k < 32; ++k) mn = fmaxf(mn, Sb[lane][k]);
      float al = __expf(mo - mn), sum = 0.0f;
#pragma unroll
      for (int k = 0; k < 32; ++k) {
        float pv = __expf(Sb[lane][k] - mn);
        Pb[lane][k] = (_Float16)pv;
        sum += pv;
      }
      mrow[lane] = mn; arow[lane] = al;
      lrow[lane] = lrow[lane] * al + sum;
    }
    __syncthreads();

    // rescale O accumulators by alpha(row)
#pragma unroll
    for (int i = 0; i < 8; ++i) {
      float ai = arow[rb + i];
#pragma unroll
      for (int dc = 0; dc < 4; ++dc) o[dc][i] *= ai;
    }

    // O += P V  (P: 16x32 A-frag from LDS; V^T gives contiguous B-frags)
    v16h ap = frag_a_at(ppb);
#pragma unroll
    for (int dc = 0; dc < 4; ++dc) {
      v16h bv = frag_b_at(pvb + (long)(dc * 16) * Tv + j);
      o[dc] = WMMA16(ap, bv, o[dc]);
    }
    __syncthreads();
  }

  // normalize by row sum and store to att[b][t][h*64+d] (f16 for the Wo GEMM)
#pragma unroll
  for (int dc = 0; dc < 4; ++dc)
#pragma unroll
    for (int i = 0; i < 8; ++i) {
      float val = o[dc][i] / lrow[rb + i];
      long t = q0 + rb + i;
      long d = dc * 16 + col;
      att[((long)b * Tv + t) * Dm + h * HD + d] = (_Float16)val;
    }
}

// ---------------- Host launcher ----------------
extern "C" void kernel_launch(void* const* d_in, const int* in_sizes, int n_in,
                              void* d_out, int out_size, void* d_ws, size_t ws_size,
                              hipStream_t stream) {
  const float* x    = (const float*)d_in[0];
  const float* cosv = (const float*)d_in[1];
  const float* sinv = (const float*)d_in[2];
  const float* Wq   = (const float*)d_in[3];
  const float* Wkv  = (const float*)d_in[4];
  const float* Wo   = (const float*)d_in[5];
  float* out = (float*)d_out;

  // workspace carve-up (all f16)
  char* ws = (char*)d_ws;
  size_t off = 0;
  auto carve = [&](size_t bytes) { void* p = ws + off; off += (bytes + 255) & ~size_t(255); return p; };
  _Float16* xh   = (_Float16*)carve((size_t)Mrows * Dm * 2);
  _Float16* WqT  = (_Float16*)carve((size_t)Dm * Dm * 2);
  _Float16* WkvT = (_Float16*)carve((size_t)KVD * Dm * 2);
  _Float16* WoT  = (_Float16*)carve((size_t)Dm * Dm * 2);
  _Float16* q16  = (_Float16*)carve((size_t)Mrows * Dm * 2);
  _Float16* kv16 = (_Float16*)carve((size_t)Mrows * KVD * 2);
  _Float16* qr   = (_Float16*)carve((size_t)Bv * HQ * Tv * HD * 2);
  _Float16* kr   = (_Float16*)carve((size_t)Bv * HKV * Tv * HD * 2);
  _Float16* vt   = (_Float16*)carve((size_t)Bv * HKV * HD * Tv * 2);
  _Float16* att  = (_Float16*)carve((size_t)Mrows * Dm * 2);

  const int TPB = 256;
  // 1) convert x, transpose+convert weights
  {
    int n = Mrows * Dm;
    k_f32_to_f16<<<(n + TPB - 1) / TPB, TPB, 0, stream>>>(x, xh, n);
    k_transpose_w<<<(Dm * Dm + TPB - 1) / TPB, TPB, 0, stream>>>(Wq, WqT, Dm, Dm);
    k_transpose_w<<<(Dm * KVD + TPB - 1) / TPB, TPB, 0, stream>>>(Wkv, WkvT, Dm, KVD);
    k_transpose_w<<<(Dm * Dm + TPB - 1) / TPB, TPB, 0, stream>>>(Wo, WoT, Dm, Dm);
  }
  // 2) projections: Q = x Wq (8192x1024x1024), KV = x Wkv (8192x1024x512)
  {
    long stripsQ = (long)(Mrows / 32) * (Dm / 64);
    k_gemm_wmma<<<(int)((stripsQ + 7) / 8), TPB, 0, stream>>>(xh, WqT, q16, nullptr,
                                                              Mrows, Dm, Dm, 1);
    long stripsKV = (long)(Mrows / 32) * (KVD / 64);
    k_gemm_wmma<<<(int)((stripsKV + 7) / 8), TPB, 0, stream>>>(xh, WkvT, kv16, nullptr,
                                                               Mrows, KVD, Dm, 1);
  }
  // 3) RoPE + head-major layouts (V transposed for contiguous B-fragments)
  {
    int nq = Bv * Tv * HQ * (HD / 2);
    k_rope_q<<<(nq + TPB - 1) / TPB, TPB, 0, stream>>>(q16, cosv, sinv, qr);
    int nkv = Bv * Tv * HKV * (HD / 2);
    k_rope_kv<<<(nkv + TPB - 1) / TPB, TPB, 0, stream>>>(kv16, cosv, sinv, kr, vt);
  }
  // 4) flash attention: one wave per (b, h, 16-query tile)
  {
    int blocks = Bv * HQ * (Tv / 16);   // 8192
    k_flash_attn<<<blocks, 32, 0, stream>>>(qr, kr, vt, att);
  }
  // 5) output projection: out = att Wo (f32 output)
  {
    long strips = (long)(Mrows / 32) * (Dm / 64);
    k_gemm_wmma<<<(int)((strips + 7) / 8), TPB, 0, stream>>>(att, WoT, nullptr, out,
                                                             Mrows, Dm, Dm, 0);
  }
}
